// GraphEncoder_17428977287344
// MI455X (gfx1250) — matrix operand order repaired and checked by reading, not compiled, not run on hardware
//
#include <hip/hip_runtime.h>
#include <hip/hip_bf16.h>
#include <math.h>

// ---------------------------------------------------------------------------
// GraphEncoder forward for MI455X (gfx1250, wave32).
//  - Dense GEMMs (encoder, per-relation RGAT transform, decoder) use
//    v_wmma_f32_16x16x32_bf16, 64x64x32 block tiles.
//    A tile staged to LDS via TDM (tensor_load_to_lds + s_wait_tensorcnt),
//    B tile staged pre-swizzled into WMMA fragment order so fragment reads
//    are 2x ds_load_b128 per 16x16 tile.
//  - Edge attention: wave-per-edge, shfl_xor reductions, atomicMax segment max
//    (monotonic uint mapping), atomicAdd segment sum + message scatter.
//  - GraphNorm: block-per-graph. TopK: bitonic sort in LDS.
//  - Set transformer (tiny): VALU, bf16 K/V in LDS, online softmax.
// ---------------------------------------------------------------------------

typedef __attribute__((ext_vector_type(16))) __bf16 v16bf;
typedef __attribute__((ext_vector_type(8)))  float  v8f;
typedef __attribute__((ext_vector_type(4)))  unsigned v4u;
typedef __attribute__((ext_vector_type(4)))  int      v4i;
typedef __attribute__((ext_vector_type(8)))  int      v8i;

#if defined(__has_builtin)
#if __has_builtin(__builtin_amdgcn_tensor_load_to_lds)
#define HAVE_TDM 1
#endif
#endif
#ifndef HAVE_TDM
#define HAVE_TDM 0
#endif

static constexpr int cN   = 16384;
static constexpr int cG   = 64;
static constexpr int cNPG = 256;
static constexpr int cE   = 131072;
static constexpr int cFIN = 128;
static constexpr int cHID = 64;
static constexpr int cD   = 512;   // gnn_output_dim
static constexpr int cR   = 8;
static constexpr int cNB  = 4;
static constexpr int cL   = 3;
static constexpr int cK   = 128;   // kept nodes per graph
static constexpr int cS   = 4;     // seeds
static constexpr int cHO  = 256;   // H*HID
static constexpr float cNEG = 0.1f;
static constexpr float cEPS = 1e-5f;

// ---------------- helpers ----------------
__device__ __forceinline__ unsigned short f2bf(float f) {
  unsigned u = __float_as_uint(f);
  return (unsigned short)((u + 0x7FFFu + ((u >> 16) & 1u)) >> 16);
}
__device__ __forceinline__ float bf2f(unsigned short h) {
  return __uint_as_float(((unsigned)h) << 16);
}
// monotonic float -> uint mapping for atomicMax-based segment max
__device__ __forceinline__ unsigned fmap(float f) {
  unsigned u = __float_as_uint(f);
  return (u & 0x80000000u) ? ~u : (u | 0x80000000u);
}
__device__ __forceinline__ float funmap(unsigned u) {
  return __uint_as_float((u & 0x80000000u) ? (u & 0x7FFFFFFFu) : ~u);
}
__device__ __forceinline__ float leaky(float v) { return v > 0.f ? v : v * cNEG; }

#if HAVE_TDM
// Issue one 2D TDM load: tile_h rows x tile_w elems (2-byte), row stride
// `stride` elems, from global `gaddr` into LDS at byte offset lds_addr.
// Descriptor bit layout per CDNA5 ISA 8.3/8.4 (D# group0 + group1).
__device__ __forceinline__ void tdm_load_2d(unsigned lds_addr, const void* gaddr,
                                            int tile_w, int tile_h, int stride) {
  unsigned long long ga = (unsigned long long)gaddr;
  v4u g0;
  g0.x = 0x1u;                                        // count=1, user descriptor
  g0.y = lds_addr;                                    // lds_addr
  g0.z = (unsigned)(ga & 0xFFFFFFFFu);                // global_addr[31:0]
  g0.w = (unsigned)((ga >> 32) & 0x01FFFFFFu) | 0x80000000u;  // [56:32] | type=2
  unsigned td0 = (unsigned)stride;                    // tensor_dim0 (elems)
  unsigned td1 = 0x7FFFFFFFu;                         // tensor_dim1: huge (no OOB)
  v8i g1;
  g1[0] = (int)(0x1u << 16);                          // data_size=1 (2 bytes)
  g1[1] = (int)((td0 & 0xFFFFu) << 16);               // tensor_dim0 lo16 @ [63:48]
  g1[2] = (int)((td0 >> 16) | ((td1 & 0xFFFFu) << 16));   // td0 hi | td1 lo
  g1[3] = (int)((td1 >> 16) | ((unsigned)tile_w << 16));  // td1 hi | tile_dim0
  g1[4] = (int)((unsigned)tile_h);                    // tile_dim1 (tile_dim2=0)
  g1[5] = (int)stride;                                // tensor_dim0_stride lo32
  g1[6] = 0;
  g1[7] = 0;
  v4i zz = {0, 0, 0, 0};
#if __clang_major__ >= 23
  v8i zz8 = {0, 0, 0, 0, 0, 0, 0, 0};
  __builtin_amdgcn_tensor_load_to_lds(g0, g1, zz, zz, zz8, 0);
#else
  __builtin_amdgcn_tensor_load_to_lds(g0, g1, zz, zz, 0);
#endif
}
#endif

// ---------------- elementwise kernels ----------------
__global__ void k_f32_to_bf16(const float* __restrict__ s, unsigned short* __restrict__ d,
                              long long n) {
  for (long long i = blockIdx.x * (long long)blockDim.x + threadIdx.x; i < n;
       i += (long long)gridDim.x * blockDim.x)
    d[i] = f2bf(s[i]);
}

__global__ void k_zero(float* __restrict__ p, long long n) {
  for (long long i = blockIdx.x * (long long)blockDim.x + threadIdx.x; i < n;
       i += (long long)gridDim.x * blockDim.x)
    p[i] = 0.f;
}

__global__ void k_bias_add(float* __restrict__ o, const float* __restrict__ bias, long long n) {
  for (long long i = blockIdx.x * (long long)blockDim.x + threadIdx.x; i < n;
       i += (long long)gridDim.x * blockDim.x)
    o[i] += bias[(int)(i & (cD - 1))];
}

__global__ void k_residual_leaky(float* __restrict__ h, const float* __restrict__ t,
                                 long long n) {
  for (long long i = blockIdx.x * (long long)blockDim.x + threadIdx.x; i < n;
       i += (long long)gridDim.x * blockDim.x)
    h[i] = leaky(t[i] + h[i]);
}

// ---------------- WMMA bf16 GEMM: C[M,N] = act(A[M,K] * B[K,N] + bias) ------
// block = 128 threads (4 waves); block tile 64x64, k-step 32.
// A staged row-major (TDM); B staged pre-swizzled into fragment order:
//   Bsw[(nt*32 + lane)*16 + h] = B[k0 + ((lane<16)?0:16) + h][nbase + nt*16 + (lane&15)]
__device__ __forceinline__ void gemm_tile(const unsigned short* __restrict__ A, int lda,
                                          const unsigned short* __restrict__ B, int ldb,
                                          float* __restrict__ C, int ldc, int K,
                                          int mbase, int nbase,
                                          const float* __restrict__ bias, int act) {
  __shared__ unsigned short As[64 * 32];
  __shared__ unsigned short Bsw[4 * 32 * 16];
  const int tid = threadIdx.x, wave = tid >> 5, lane = tid & 31;
  // A coop-load fallback mapping (when no TDM)
  const int ra = tid >> 1, ca = (tid & 1) * 16;
  // B coop-load mapping: row pair 2*rp,2*rp+1, 8-col group cg
  const int rp = tid >> 3, cg = tid & 7;
  const int r0 = 2 * rp;
  const int ntb = cg >> 1;
  const int lbase = (cg & 1) * 8 + ((r0 >= 16) ? 16 : 0);
  const int hh0 = r0 & 15;
  v8f acc[4] = {};
  for (int k0 = 0; k0 < K; k0 += 32) {
#if HAVE_TDM
    if (wave == 0) {  // wave-uniform branch: only wave 0 issues the DMA
      tdm_load_2d((unsigned)(unsigned long long)(const void*)As,
                  (const void*)(A + (size_t)mbase * lda + k0), 32, 64, lda);
    }
#else
    *reinterpret_cast<uint4*>(&As[ra * 32 + ca]) =
        *reinterpret_cast<const uint4*>(A + (size_t)(mbase + ra) * lda + k0 + ca);
#endif
    // B tile -> registers -> swizzled LDS (pairs of K packed as b32 stores)
    {
      uint4 row0 = *reinterpret_cast<const uint4*>(B + (size_t)(k0 + r0) * ldb + nbase + cg * 8);
      uint4 row1 = *reinterpret_cast<const uint4*>(B + (size_t)(k0 + r0 + 1) * ldb + nbase + cg * 8);
      const unsigned short* u0 = (const unsigned short*)&row0;
      const unsigned short* u1 = (const unsigned short*)&row1;
#pragma unroll
      for (int i = 0; i < 8; ++i) {
        unsigned pack = (unsigned)u0[i] | ((unsigned)u1[i] << 16);
        *reinterpret_cast<unsigned*>(&Bsw[(ntb * 32 + lbase + i) * 16 + hh0]) = pack;
      }
      if (k0 + 32 < K)
        __builtin_prefetch(B + (size_t)(k0 + 32 + r0) * ldb + nbase + cg * 8, 0, 1);
    }
#if HAVE_TDM
    if (wave == 0) __builtin_amdgcn_s_wait_tensorcnt(0);
#endif
    __syncthreads();
    // A fragment (16x32 bf16): per lane two contiguous 16B chunks
    union { v16bf v; unsigned short u[16]; } af;
    {
      const int m = lane & 15, kb = (lane < 16) ? 0 : 16;
      const unsigned short* ap = &As[(wave * 16 + m) * 32 + kb];
      *reinterpret_cast<uint4*>(&af.u[0]) = *reinterpret_cast<const uint4*>(ap);
      *reinterpret_cast<uint4*>(&af.u[8]) = *reinterpret_cast<const uint4*>(ap + 8);
    }
    // B fragments: 32 contiguous bytes per lane per 16x16 tile
#pragma unroll
    for (int nt = 0; nt < 4; ++nt) {
      union { v16bf v; unsigned short u[16]; } bfr;
      const unsigned short* bp = &Bsw[(nt * 32 + lane) * 16];
      *reinterpret_cast<uint4*>(&bfr.u[0]) = *reinterpret_cast<const uint4*>(bp);
      *reinterpret_cast<uint4*>(&bfr.u[8]) = *reinterpret_cast<const uint4*>(bp + 8);
      acc[nt] = __builtin_amdgcn_wmma_f32_16x16x32_bf16(
          false, af.v, false, bfr.v, (short)0, acc[nt], false, false);
    }
    __syncthreads();
  }
  // C layout: lane -> col = lane&15; VGPR h -> row h (lanes 0-15) / 8+h (lanes 16-31)
  const int n = lane & 15, rowoff = (lane < 16) ? 0 : 8;
#pragma unroll
  for (int nt = 0; nt < 4; ++nt) {
    int col = nbase + nt * 16 + n;
    float bv = bias ? bias[col] : 0.f;
#pragma unroll
    for (int hh = 0; hh < 8; ++hh) {
      int row = mbase + wave * 16 + rowoff + hh;
      float v = acc[nt][hh] + bv;
      if (act) v = leaky(v);
      C[(size_t)row * ldc + col] = v;
    }
  }
  (void)ra; (void)ca;
}

__global__ void k_gemm_bf16(const unsigned short* __restrict__ A, int lda,
                            const unsigned short* __restrict__ B, int ldb,
                            float* __restrict__ C, int ldc, int K,
                            const float* __restrict__ bias, int act) {
  gemm_tile(A, lda, B, ldb, C, ldc, K, blockIdx.y * 64, blockIdx.x * 64, bias, act);
}

// per-relation block-diagonal transform: t[r,n,b*64+f] = h[n, b*128:] @ W[r,b]
__global__ void k_gemm_rel(const unsigned short* __restrict__ h16,
                           const unsigned short* __restrict__ w16,
                           float* __restrict__ t) {
  const int z = blockIdx.z, r = z >> 2, b = z & 3;
  const unsigned short* A = h16 + b * 128;                 // lda = cD
  const unsigned short* B = w16 + (size_t)z * 128 * 64;    // ldb = 64
  float* C = t + (size_t)r * cN * cHO + b * 64;            // ldc = cHO
  gemm_tile(A, cD, B, 64, C, cHO, 128, blockIdx.y * 64, blockIdx.x * 64, nullptr, 0);
}

// ---------------- GraphNorm (block per graph) ----------------
__global__ void k_graph_norm(float* __restrict__ x, const float* __restrict__ w,
                             const float* __restrict__ b, const float* __restrict__ msc) {
  const int g = blockIdx.x, tid = threadIdx.x;
  for (int f = tid; f < cD; f += 256) {
    float s = 0.f;
    for (int n = 0; n < cNPG; ++n) s += x[((size_t)g * cNPG + n) * cD + f];
    float ms = (s * (1.f / cNPG)) * msc[f];
    float var = 0.f;
    for (int n = 0; n < cNPG; ++n) {
      float o = x[((size_t)g * cNPG + n) * cD + f] - ms;
      var += o * o;
    }
    var *= (1.f / cNPG);
    float inv = rsqrtf(var + cEPS);
    float wf = w[f], bf = b[f];
    for (int n = 0; n < cNPG; ++n) {
      size_t idx = ((size_t)g * cNPG + n) * cD + f;
      x[idx] = wf * (x[idx] - ms) * inv + bf;
    }
  }
}

// ---------------- edge-side kernels ----------------
__global__ void k_etype(const float* __restrict__ ea, int* __restrict__ et) {
  int e = blockIdx.x * 256 + threadIdx.x;
  if (e >= cE) return;
  float best = ea[(size_t)e * 8];
  int bi = 0;
  for (int j = 1; j < 8; ++j) {
    float v = ea[(size_t)e * 8 + j];
    if (v > best) { best = v; bi = j; }
  }
  et[e] = bi;
}

__global__ void k_ae8(const float* __restrict__ lin_edge /*8x256*/,
                      const float* __restrict__ e /*256x8*/, float* __restrict__ ae8) {
  int tid = threadIdx.x;  // 64 threads
  int m = tid >> 3, c = tid & 7;
  float s = 0.f;
  for (int d = 0; d < cHO; ++d) s += lin_edge[m * cHO + d] * e[d * 8 + c];
  ae8[m * 8 + c] = s;
}

__global__ void k_init_seg(unsigned* __restrict__ segm, float* __restrict__ segs) {
  int i = blockIdx.x * 256 + threadIdx.x;
  if (i < cN * 8) { segm[i] = 0x007FFFFFu; /* fmap(-inf) */ segs[i] = 0.f; }
}

// wave-per-edge: a[e,c] = (ti@q)[c] * (tj@k)[c] * (eattr@ae8)[c]; segment max
__global__ void k_edge_logits(const float* __restrict__ t, const float* __restrict__ q,
                              const float* __restrict__ kmat, const float* __restrict__ ae8,
                              const float* __restrict__ eattr,
                              const int* __restrict__ src, const int* __restrict__ dst,
                              const int* __restrict__ etype,
                              float* __restrict__ abuf, unsigned* __restrict__ segm) {
  const int e = blockIdx.x * 4 + (threadIdx.x >> 5);
  const int lane = threadIdx.x & 31;
  if (e >= cE) return;
  const int r = etype[e], sN = src[e], dN = dst[e];
  const float* rowi = t + ((size_t)r * cN + dN) * cHO;
  const float* rowj = t + ((size_t)r * cN + sN) * cHO;
  float qi[8] = {0, 0, 0, 0, 0, 0, 0, 0};
  float kj[8] = {0, 0, 0, 0, 0, 0, 0, 0};
  for (int dd = 0; dd < 8; ++dd) {
    int idx = lane * 8 + dd;
    float xi = rowi[idx], xj = rowj[idx];
#pragma unroll
    for (int c = 0; c < 8; ++c) {
      qi[c] += xi * q[idx * 8 + c];
      kj[c] += xj * kmat[idx * 8 + c];
    }
  }
#pragma unroll
  for (int c = 0; c < 8; ++c)
#pragma unroll
    for (int off = 16; off; off >>= 1) {
      qi[c] += __shfl_xor(qi[c], off, 32);
      kj[c] += __shfl_xor(kj[c], off, 32);
    }
  if (lane < 8) {
    const int c = lane;
    float ae = 0.f;
#pragma unroll
    for (int m = 0; m < 8; ++m) ae += eattr[(size_t)e * 8 + m] * ae8[m * 8 + c];
    float a = qi[c] * kj[c] * ae;
    abuf[(size_t)e * 8 + c] = a;
    atomicMax(&segm[(size_t)dN * 8 + c], fmap(a));
  }
}

__global__ void k_edge_exp(float* __restrict__ abuf, const unsigned* __restrict__ segm,
                           float* __restrict__ segs, const int* __restrict__ dst) {
  int tid = blockIdx.x * 256 + threadIdx.x;  // E*8
  if (tid >= cE * 8) return;
  int e = tid >> 3, c = tid & 7, dN = dst[e];
  float m = funmap(segm[(size_t)dN * 8 + c]);
  float ex = __expf(abuf[tid] - m);
  abuf[tid] = ex;
  atomicAdd(&segs[(size_t)dN * 8 + c], ex);
}

__global__ void k_edge_msg(const float* __restrict__ abuf, const float* __restrict__ segs,
                           const float* __restrict__ t,
                           const int* __restrict__ src, const int* __restrict__ dst,
                           const int* __restrict__ etype, float* __restrict__ out) {
  long long tid = blockIdx.x * 256LL + threadIdx.x;  // E*64
  int e = (int)(tid >> 6), hid = (int)(tid & 63);
  int r = etype[e], sN = src[e], dN = dst[e];
  const float* rowj = t + ((size_t)r * cN + sN) * cHO;
  float tj[4];
#pragma unroll
  for (int h = 0; h < 4; ++h) tj[h] = rowj[h * 64 + hid];
#pragma unroll
  for (int c = 0; c < 8; ++c) {
    float al = abuf[(size_t)e * 8 + c] / (segs[(size_t)dN * 8 + c] + 1e-16f);
    atomicAdd(&out[(size_t)dN * cD + c * 64 + hid], al * tj[c >> 1]);
  }
}

// ---------------- pooling ----------------
__global__ void k_pool_agg(const float* __restrict__ hdec, const int* __restrict__ src,
                           const int* __restrict__ dst, float* __restrict__ agg) {
  long long tid = blockIdx.x * 256LL + threadIdx.x;  // E*64
  int e = (int)(tid >> 6), j = (int)(tid & 63);
  atomicAdd(&agg[(size_t)dst[e] * cHID + j], hdec[(size_t)src[e] * cHID + j]);
}

__global__ void k_score(const float* __restrict__ hdec, const float* __restrict__ agg,
                        const float* __restrict__ relw, const float* __restrict__ relb,
                        const float* __restrict__ rootw, float* __restrict__ score) {
  int n = blockIdx.x * 256 + threadIdx.x;
  if (n >= cN) return;
  float s = relb[0];
  for (int j = 0; j < cHID; ++j)
    s += agg[(size_t)n * cHID + j] * relw[j] + hdec[(size_t)n * cHID + j] * rootw[j];
  score[n] = tanhf(s);
}

// per-graph top-128 (bitonic sort on 256 elems, descending, stable by index)
__global__ void k_topk_xs(const float* __restrict__ score, const float* __restrict__ hdec,
                          float* __restrict__ xs, float* __restrict__ out_tail) {
  __shared__ float ks[256];
  __shared__ int ki[256];
  const int g = blockIdx.x, tid = threadIdx.x;
  ks[tid] = score[g * cNPG + tid];
  ki[tid] = tid;
  for (int k = 2; k <= 256; k <<= 1)
    for (int j = k >> 1; j > 0; j >>= 1) {
      __syncthreads();
      int ixj = tid ^ j;
      if (ixj > tid) {
        float a = ks[tid], b = ks[ixj];
        int ia = ki[tid], ib = ki[ixj];
        bool desc = ((tid & k) == 0);
        bool sw = desc ? (a < b || (a == b && ia > ib)) : (a > b || (a == b && ia < ib));
        if (sw) { ks[tid] = b; ks[ixj] = a; ki[tid] = ib; ki[ixj] = ia; }
      }
    }
  __syncthreads();
  if (tid < cK) {
    int node = g * cNPG + ki[tid];
    float v = ks[tid];
    for (int j = 0; j < cHID; ++j)
      xs[((size_t)g * cK + tid) * cHID + j] = hdec[(size_t)node * cHID + j] * v;
    out_tail[g * cK + tid] = (float)g;  // pooled_batch
  }
}

// ---------------- set transformer (MAB) ----------------
// out = h + relu(h @ lin_w.T + lin_b),  h = mha(Xq, Ykv) + Xq.  c=64, heads=4, dh=16.
__global__ void k_mab(const float* __restrict__ Xq, long long xqStride, int Lq,
                      const float* __restrict__ Y, long long yStride, int Lk,
                      const float* __restrict__ in_w, const float* __restrict__ in_b,
                      const float* __restrict__ out_w, const float* __restrict__ out_b,
                      const float* __restrict__ lin_w, const float* __restrict__ lin_b,
                      float* __restrict__ Out, long long outStride, int nanClean) {
  __shared__ unsigned short kb[128 * 64];
  __shared__ unsigned short vb[128 * 64];
  const int g = blockIdx.x, tid = threadIdx.x;
  // cooperative K,V = Y@Wk.T+bk, Y@Wv.T+bv  (bf16 in LDS)
  for (int idx = tid; idx < Lk * 64; idx += 128) {
    int j = idx >> 6, cc = idx & 63;
    const float* yrow = Y + (size_t)g * yStride + (size_t)j * 64;
    float sk = in_b[64 + cc], sv = in_b[128 + cc];
    for (int d = 0; d < 64; ++d) {
      float yd = yrow[d];
      sk += yd * in_w[(64 + cc) * 64 + d];
      sv += yd * in_w[(128 + cc) * 64 + d];
    }
    kb[idx] = f2bf(sk);
    vb[idx] = f2bf(sv);
  }
  __syncthreads();
  const int i = tid;
  if (i >= Lq) return;
  const float* xrow = Xq + (size_t)g * xqStride + (size_t)i * 64;
  float qrow[64];
  for (int cc = 0; cc < 64; ++cc) {
    float s = in_b[cc];
    for (int d = 0; d < 64; ++d) s += xrow[d] * in_w[cc * 64 + d];
    qrow[cc] = s;
  }
  float orow[64];
  for (int hd = 0; hd < 4; ++hd) {
    float mx = -1e30f, ssum = 0.f, acc[16];
#pragma unroll
    for (int d = 0; d < 16; ++d) acc[d] = 0.f;
    for (int j = 0; j < Lk; ++j) {
      float logit = 0.f;
#pragma unroll
      for (int d = 0; d < 16; ++d) logit += qrow[hd * 16 + d] * bf2f(kb[j * 64 + hd * 16 + d]);
      logit *= 0.25f;  // 1/sqrt(16)
      if (logit > mx) {
        float corr = __expf(mx - logit);
        ssum *= corr;
#pragma unroll
        for (int d = 0; d < 16; ++d) acc[d] *= corr;
        mx = logit;
      }
      float w = __expf(logit - mx);
      ssum += w;
#pragma unroll
      for (int d = 0; d < 16; ++d) acc[d] += w * bf2f(vb[j * 64 + hd * 16 + d]);
    }
    float inv = 1.f / ssum;
#pragma unroll
    for (int d = 0; d < 16; ++d) orow[hd * 16 + d] = acc[d] * inv;
  }
  float outrow[64];
  for (int cc = 0; cc < 64; ++cc) {
    float s = out_b[cc];
    for (int d = 0; d < 64; ++d) s += orow[d] * out_w[cc * 64 + d];
    outrow[cc] = s + xrow[cc];
  }
  for (int cc = 0; cc < 64; ++cc) {
    float s = lin_b[cc];
    for (int d = 0; d < 64; ++d) s += outrow[d] * lin_w[cc * 64 + d];
    float v = outrow[cc] + (s > 0.f ? s : 0.f);
    if (nanClean) {
      if (isnan(v)) v = 0.f;
      if (v > 3.4028235e38f) v = 3.4028235e38f;
      if (v < -3.4028235e38f) v = -3.4028235e38f;
    }
    Out[(size_t)g * outStride + (size_t)i * 64 + cc] = v;
  }
}

__global__ void k_kv_relu(const float* __restrict__ z, const float* __restrict__ w,
                          const float* __restrict__ b, float* __restrict__ kv) {
  int idx = blockIdx.x * 256 + threadIdx.x;  // G*128*64
  if (idx >= cG * cK * cHID) return;
  int row = idx >> 6, cc = idx & 63;
  float s = b[cc];
  for (int d = 0; d < 64; ++d) s += z[(size_t)row * 64 + d] * w[cc * 64 + d];
  kv[idx] = s > 0.f ? s : 0.f;
}

// ---------------------------------------------------------------------------
// Input flattening (jax tree: dict keys alphabetical):
//  0 x, 1 edge_index, 2 edge_attr, 3 batch,
//  4+6l: convs[l]{bias,e,k,lin_edge,q,weight}, 22 dec_b,
//  23..28 dec_sab{in_b,in_w,lin_b,lin_w,out_b,out_w}, 29 dec_w, 30 enc_b,
//  31..36 enc_sab{...}, 37 enc_w, 38..40 gn_in{bias,mean_scale,weight},
//  41+3l gns[l]{bias,mean_scale,weight}, 50 pma_lin_b, 51 pma_lin_w,
//  52..57 pma_mab{...}, 58 pma_seed, 59 pool_rel_b, 60 pool_rel_w, 61 pool_root_w
// ---------------------------------------------------------------------------
extern "C" void kernel_launch(void* const* d_in, const int* in_sizes, int n_in,
                              void* d_out, int out_size, void* d_ws, size_t ws_size,
                              hipStream_t stream) {
  (void)in_sizes; (void)n_in; (void)out_size; (void)ws_size;
  const float* x     = (const float*)d_in[0];
  const int*   ei    = (const int*)d_in[1];
  const int*   srcp  = ei;
  const int*   dstp  = ei + cE;
  const float* eattr = (const float*)d_in[2];

  // workspace layout (~233 MiB)
  char* ws = (char*)d_ws;
  size_t off = 0;
  auto take = [&](size_t bytes) -> char* {
    char* p = ws + off;
    off = (off + bytes + 255) & ~(size_t)255;
    return p;
  };
  float* h              = (float*)take((size_t)cN * cD * 4);
  float* tmp            = (float*)take((size_t)cN * cD * 4);
  unsigned short* h16   = (unsigned short*)take((size_t)cN * cD * 2);
  unsigned short* x16   = (unsigned short*)take((size_t)cN * cFIN * 2);
  float* t              = (float*)take((size_t)cR * cN * cHO * 4);
  unsigned short* encw16  = (unsigned short*)take((size_t)cFIN * cD * 2);
  unsigned short* decw16  = (unsigned short*)take((size_t)cD * cHID * 2);
  unsigned short* convw16 = (unsigned short*)take((size_t)cL * cR * cNB * 128 * 64 * 2);
  float* abuf   = (float*)take((size_t)cE * 8 * 4);
  unsigned* segm = (unsigned*)take((size_t)cN * 8 * 4);
  float* segs   = (float*)take((size_t)cN * 8 * 4);
  int* etype    = (int*)take((size_t)cE * 4);
  float* ae8    = (float*)take(64 * 4);
  float* hdec   = (float*)take((size_t)cN * cHID * 4);
  float* agg    = (float*)take((size_t)cN * cHID * 4);
  float* score  = (float*)take((size_t)cN * 4);
  float* xs     = (float*)take((size_t)cG * cK * cHID * 4);
  float* z      = (float*)take((size_t)cG * cK * cHID * 4);
  float* kvbuf  = (float*)take((size_t)cG * cK * cHID * 4);
  float* z2     = (float*)take((size_t)cG * cS * cHID * 4);

  auto cvt = [&](const float* s, unsigned short* dp, long long n) {
    int blocks = (int)((n + 255) / 256);
    k_f32_to_bf16<<<dim3(blocks), dim3(256), 0, stream>>>(s, dp, n);
  };

  // --- bf16 conversions of GEMM operands ---
  cvt(x, x16, (long long)cN * cFIN);
  cvt((const float*)d_in[37], encw16, (long long)cFIN * cD);
  cvt((const float*)d_in[29], decw16, (long long)cD * cHID);
  for (int l = 0; l < cL; ++l)
    cvt((const float*)d_in[4 + l * 6 + 5], convw16 + (size_t)l * cR * cNB * 128 * 64,
        (long long)cR * cNB * 128 * 64);

  // --- encoder: h = leaky(x @ enc_w + enc_b); graph_norm ---
  k_gemm_bf16<<<dim3(cD / 64, cN / 64), dim3(128), 0, stream>>>(
      x16, cFIN, encw16, cD, h, cD, cFIN, (const float*)d_in[30], 1);
  k_graph_norm<<<dim3(cG), dim3(256), 0, stream>>>(
      h, (const float*)d_in[40], (const float*)d_in[38], (const float*)d_in[39]);
  k_etype<<<dim3(cE / 256), dim3(256), 0, stream>>>(eattr, etype);

  // --- RGAT layers ---
  for (int l = 0; l < cL; ++l) {
    const float* cbias = (const float*)d_in[4 + l * 6 + 0];
    const float* ce    = (const float*)d_in[4 + l * 6 + 1];
    const float* ck    = (const float*)d_in[4 + l * 6 + 2];
    const float* cle   = (const float*)d_in[4 + l * 6 + 3];
    const float* cq    = (const float*)d_in[4 + l * 6 + 4];

    cvt(h, h16, (long long)cN * cD);
    k_gemm_rel<<<dim3(1, cN / 64, cR * cNB), dim3(128), 0, stream>>>(
        h16, convw16 + (size_t)l * cR * cNB * 128 * 64, t);
    k_ae8<<<dim3(1), dim3(64), 0, stream>>>(cle, ce, ae8);
    k_init_seg<<<dim3(cN * 8 / 256), dim3(256), 0, stream>>>(segm, segs);
    k_edge_logits<<<dim3(cE / 4), dim3(128), 0, stream>>>(
        t, cq, ck, ae8, eattr, srcp, dstp, etype, abuf, segm);
    k_edge_exp<<<dim3(cE * 8 / 256), dim3(256), 0, stream>>>(abuf, segm, segs, dstp);
    k_zero<<<dim3(8192), dim3(256), 0, stream>>>(tmp, (long long)cN * cD);
    k_edge_msg<<<dim3(cE * 64 / 256), dim3(256), 0, stream>>>(
        abuf, segs, t, srcp, dstp, etype, tmp);
    k_bias_add<<<dim3(8192), dim3(256), 0, stream>>>(tmp, cbias, (long long)cN * cD);
    k_graph_norm<<<dim3(cG), dim3(256), 0, stream>>>(
        tmp, (const float*)d_in[41 + l * 3 + 2], (const float*)d_in[41 + l * 3 + 0],
        (const float*)d_in[41 + l * 3 + 1]);
    k_residual_leaky<<<dim3(8192), dim3(256), 0, stream>>>(h, tmp, (long long)cN * cD);
  }

  // --- decoder: hdec = leaky(h @ dec_w + dec_b) ---
  cvt(h, h16, (long long)cN * cD);
  k_gemm_bf16<<<dim3(cHID / 64, cN / 64), dim3(128), 0, stream>>>(
      h16, cD, decw16, cHID, hdec, cHID, cD, (const float*)d_in[22], 1);

  // --- SAGPool score + top-k ---
  k_zero<<<dim3(4096), dim3(256), 0, stream>>>(agg, (long long)cN * cHID);
  k_pool_agg<<<dim3(cE * 64 / 256), dim3(256), 0, stream>>>(hdec, srcp, dstp, agg);
  k_score<<<dim3(cN / 256), dim3(256), 0, stream>>>(
      hdec, agg, (const float*)d_in[60], (const float*)d_in[59], (const float*)d_in[61],
      score);
  k_topk_xs<<<dim3(cG), dim3(256), 0, stream>>>(score, hdec, xs,
                                                (float*)d_out + cG * cS * cHID);

  // --- Set transformer: SAB -> PMA -> SAB ---
  // enc_sab on xs
  k_mab<<<dim3(cG), dim3(128), 0, stream>>>(
      xs, (long long)cK * cHID, cK, xs, (long long)cK * cHID, cK,
      (const float*)d_in[32], (const float*)d_in[31], (const float*)d_in[36],
      (const float*)d_in[35], (const float*)d_in[34], (const float*)d_in[33],
      z, (long long)cK * cHID, 0);
  // kv = relu(z @ pma_lin_w.T + pma_lin_b)
  k_kv_relu<<<dim3(cG * cK * cHID / 256), dim3(256), 0, stream>>>(
      z, (const float*)d_in[51], (const float*)d_in[50], kvbuf);
  // PMA: mab(seed, kv)
  k_mab<<<dim3(cG), dim3(128), 0, stream>>>(
      (const float*)d_in[58], 0LL, cS, kvbuf, (long long)cK * cHID, cK,
      (const float*)d_in[53], (const float*)d_in[52], (const float*)d_in[57],
      (const float*)d_in[56], (const float*)d_in[55], (const float*)d_in[54],
      z2, (long long)cS * cHID, 0);
  // dec_sab: mab(z2, z2) -> d_out (with nan_to_num)
  k_mab<<<dim3(cG), dim3(128), 0, stream>>>(
      z2, (long long)cS * cHID, cS, z2, (long long)cS * cHID, cS,
      (const float*)d_in[24], (const float*)d_in[23], (const float*)d_in[28],
      (const float*)d_in[27], (const float*)d_in[26], (const float*)d_in[25],
      (float*)d_out, (long long)cS * cHID, 1);
}
